// LSTM_AE_shared_26731876450991
// MI455X (gfx1250) — compile-verified, matrix operand order checked by reading, not run on hardware
//
#include <hip/hip_runtime.h>

typedef _Float16 v16h __attribute__((ext_vector_type(16)));
typedef _Float16 v8h  __attribute__((ext_vector_type(8)));
typedef float    v8f  __attribute__((ext_vector_type(8)));
typedef float    v4f  __attribute__((ext_vector_type(4)));

#define B_   1024
#define T_   1024
#define H_   128
#define G_   512          // 4*H
#define ROWS 16           // batch rows per workgroup
#define KT   4            // H/32 K-tiles per WMMA chain
#define HSTRIDE (ROWS * H_)   // halfs per h ping-pong buffer

__device__ __forceinline__ float fast_rcp(float x) {
    return __builtin_amdgcn_rcpf(x);          // v_rcp_f32 (TRANS), no divide expansion
}
__device__ __forceinline__ float sigf(float x) {
    return fast_rcp(1.0f + __expf(-x));
}
__device__ __forceinline__ float tanh_fast(float x) {
    float e = __expf(-2.0f * x);
    return (1.0f - e) * fast_rcp(1.0f + e);
}

__global__ void __launch_bounds__(256, 1)
lstm_ae_kernel(const float* __restrict__ x,
               const float* __restrict__ W_ih_e, const float* __restrict__ W_hh_e,
               const float* __restrict__ b_ih_e, const float* __restrict__ b_hh_e,
               const float* __restrict__ W_ih_d, const float* __restrict__ W_hh_d,
               const float* __restrict__ b_ih_d, const float* __restrict__ b_hh_d,
               const float* __restrict__ W_lin,  const float* __restrict__ b_lin,
               float* __restrict__ out)
{
    extern __shared__ char smem[];
    // LDS carve-out (CDNA5: 320KB/WGP, ~141KB used)
    _Float16* wlds  = (_Float16*)smem;            // 512*128 halfs = 128KB: staging for B-fragment swizzle
    _Float16* hbuf  = wlds + G_ * H_;             // 2 * 16*128 halfs: h ping-pong (offset-selected!)
    float* wih_s    = (float*)(hbuf + 2 * HSTRIDE); // 512 floats: W_ih (staging)
    float* bias_s   = wih_s + G_;                 // 512 floats: b_ih + b_hh (staging)
    float* wlin_s   = bias_s + G_;                // 128 floats: W_lin (staging)
    float* xs2      = wlin_s + H_;                // 2*16 floats: scalar input ping-pong
    float* ypart    = xs2 + 2 * ROWS;             // 128 floats: per-wave projection partials

    const int tid  = threadIdx.x;
    const int lane = tid & 31;
    const int wv   = tid >> 5;                    // wave 0..7; owns j-block jbg = wv
    const int row0 = blockIdx.x * ROWS;

    const int m    = lane & 15;                   // A-fragment row
    const int khi  = (lane & 16) ? 8 : 0;         // A-fragment K sub-offset
    const int n    = lane & 15;                   // C/D column within tile
    const int rsel = (lane & 16) ? 8 : 0;         // C/D row group
    const int jcol = wv * 16 + n;                 // owned h-column 0..127

    float cst[8];                                 // c-state (rows rsel..rsel+7, col jcol)
    #pragma unroll
    for (int r = 0; r < 8; ++r) cst[r] = 0.0f;

    // Phase-resident registers: W_hh B fragments (128 VGPRs) + per-column constants.
    v16h  bfrag[4][KT];                           // bfrag[g][kt]: gate-tile nt = g*8 + wv
    float wih_r[4], bias_r[4], wlin_r;

    union HV { v16h v; v8h h[2]; };

    // ---- phase loader: swizzle W_hh into B-fragment order in LDS, then lift to VGPRs ----
    // tile (nt,kt): 32 lane-slots * 16 halfs; lanes 0-15: n=slot, K {0..7,16..23};
    // lanes 16-31: n=slot-16, K {8..15,24..31}.  B[k][n] = W_hh[col=n_glob][j=k_glob].
    auto load_phase = [&](const float* Whh, const float* Wih,
                          const float* bih, const float* bhh) {
        __syncthreads();
        for (int d = tid; d < G_ * H_; d += 256) {
            int tile   = d >> 9;
            int within = d & 511;
            int s  = within >> 4;
            int e  = within & 15;
            int nt = tile >> 2;
            int kt = tile & 3;
            int col = nt * 16 + (s & 15);
            int k   = kt * 32 + ((s & 16) ? 8 : 0) + ((e < 8) ? e : (e + 8));
            wlds[d] = (_Float16)Whh[col * H_ + k];
        }
        for (int c = tid; c < G_; c += 256) {
            wih_s[c]  = Wih[c];
            bias_s[c] = bih[c] + bhh[c];
        }
        __syncthreads();
        #pragma unroll
        for (int g = 0; g < 4; ++g) {
            const int nt = g * 8 + wv;
            #pragma unroll
            for (int kt = 0; kt < KT; ++kt) {
                HV b;
                const v8h* bp = (const v8h*)(wlds + (((nt << 2) + kt) << 9) + (lane << 4));
                b.h[0] = bp[0];
                b.h[1] = bp[1];
                bfrag[g][kt] = b.v;
            }
        }
        // hoist loop-invariant per-column constants into VGPRs
        #pragma unroll
        for (int g = 0; g < 4; ++g) {
            wih_r[g]  = wih_s[g * H_ + jcol];
            bias_r[g] = bias_s[g * H_ + jcol];
        }
        wlin_r = wlin_s[jcol];
    };

    // ---- one LSTM cell step; ping-pong selected by INTEGER OFFSETS so the
    //      compiler keeps addrspace(3) => ds_load_b128, not flat_load ----
    auto cell_step = [&](int hin_off, int hout_off, int x_off, bool doproj) {
        // A fragments of h from LDS (ISA 16-bit A 16x32 layout)
        v16h afrag[KT];
        #pragma unroll
        for (int kt = 0; kt < KT; ++kt) {
            HV a;
            const v8h* p = (const v8h*)(hbuf + hin_off + m * H_ + kt * 32 + khi);
            a.h[0] = p[0];       // K base..base+7
            a.h[1] = p[2];       // K base+16..base+23
            afrag[kt] = a.v;
        }
        // per-row scalar inputs for this lane's row group (two b128 loads)
        v4f xlo = *(const v4f*)(xs2 + x_off + rsel);
        v4f xhi = *(const v4f*)(xs2 + x_off + rsel + 4);

        const v8f zero = {0.f,0.f,0.f,0.f,0.f,0.f,0.f,0.f};
        v8f acc[4];
        #pragma unroll
        for (int g = 0; g < 4; ++g) acc[g] = zero;
        #pragma unroll
        for (int g = 0; g < 4; ++g)
            #pragma unroll
            for (int kt = 0; kt < KT; ++kt)
                acc[g] = __builtin_amdgcn_wmma_f32_16x16x32_f16(
                    false, afrag[kt], false, bfrag[g][kt], (short)0, acc[g],
                    false, false);

        // elementwise cell update (C/D layout: elem r -> row r+rsel, lane -> col n)
        float ypl[8];
        #pragma unroll
        for (int r = 0; r < 8; ++r) {
            const float xv = (r < 4) ? xlo[r] : xhi[r - 4];
            float gi = acc[0][r] + xv * wih_r[0] + bias_r[0];
            float gf = acc[1][r] + xv * wih_r[1] + bias_r[1];
            float gg = acc[2][r] + xv * wih_r[2] + bias_r[2];
            float go = acc[3][r] + xv * wih_r[3] + bias_r[3];
            float cn = sigf(gf) * cst[r] + sigf(gi) * tanh_fast(gg);
            cst[r] = cn;
            float hn = sigf(go) * tanh_fast(cn);
            hbuf[hout_off + (r + rsel) * H_ + jcol] = (_Float16)hn;
            ypl[r] = hn * wlin_r;
        }
        if (doproj) {
            // reduce over the 16 N-lanes (butterfly stays within each 16-lane half)
            #pragma unroll
            for (int r = 0; r < 8; ++r) {
                float v = ypl[r];
                v += __shfl_xor(v, 1, 32);
                v += __shfl_xor(v, 2, 32);
                v += __shfl_xor(v, 4, 32);
                v += __shfl_xor(v, 8, 32);
                ypl[r] = v;
            }
            if ((lane & 15) == 0) {
                #pragma unroll
                for (int r = 0; r < 8; ++r)
                    ypart[wv * 16 + rsel + r] = ypl[r];
            }
        }
    };

    // ---- combine projection partials: y -> xs2[xdst_off] (and optionally output) ----
    auto finalize_y = [&](bool write_out, int t, int xdst_off) {
        __syncthreads();                       // ypart visible
        if (tid < 16) {
            float yv = b_lin[0];
            #pragma unroll
            for (int w = 0; w < 8; ++w) yv += ypart[w * 16 + tid];
            if (write_out) out[(size_t)(row0 + tid) * T_ + (T_ - 1 - t)] = yv;
            xs2[xdst_off + tid] = yv;
        }
    };

    // ---- init ----
    if (tid < H_) wlin_s[tid] = W_lin[tid];
    for (int d = tid; d < HSTRIDE; d += 256) hbuf[d] = (_Float16)0.0f;
    if (tid < 16) xs2[tid] = x[(size_t)(row0 + tid) * T_ + 0];   // preload x_0
    load_phase(W_hh_e, W_ih_e, b_ih_e, b_hh_e);

    // ---- encoder: 1 barrier/step; prefetch x[:,t+1] during step t ----
    #pragma unroll 1
    for (int t = 0; t < T_ - 1; ++t) {
        __syncthreads();
        if (tid < 16) xs2[((t + 1) & 1) * ROWS + tid] =
            x[(size_t)(row0 + tid) * T_ + (t + 1)];
        cell_step((t & 1) * HSTRIDE, ((t + 1) & 1) * HSTRIDE, (t & 1) * ROWS, false);
    }
    __syncthreads();
    cell_step(((T_ - 1) & 1) * HSTRIDE, 0, ((T_ - 1) & 1) * ROWS, true);
    finalize_y(false, 0, 0);                   // y0 -> xs2 buf 0

    // ---- decoder: 2 barriers/step ----
    load_phase(W_hh_d, W_ih_d, b_ih_d, b_hh_d);
    #pragma unroll 1
    for (int t = 0; t < T_; ++t) {
        __syncthreads();
        cell_step((t & 1) * HSTRIDE, ((t + 1) & 1) * HSTRIDE, (t & 1) * ROWS, true);
        finalize_y(true, t, ((t + 1) & 1) * ROWS);   // out[:,T-1-t], y -> next xs
    }
}

extern "C" void kernel_launch(void* const* d_in, const int* in_sizes, int n_in,
                              void* d_out, int out_size, void* d_ws, size_t ws_size,
                              hipStream_t stream) {
    (void)in_sizes; (void)n_in; (void)out_size; (void)d_ws; (void)ws_size;
    const float* x     = (const float*)d_in[0];
    const float* Wih_e = (const float*)d_in[1];
    const float* Whh_e = (const float*)d_in[2];
    const float* bih_e = (const float*)d_in[3];
    const float* bhh_e = (const float*)d_in[4];
    const float* Wih_d = (const float*)d_in[5];
    const float* Whh_d = (const float*)d_in[6];
    const float* bih_d = (const float*)d_in[7];
    const float* bhh_d = (const float*)d_in[8];
    const float* Wlin  = (const float*)d_in[9];
    const float* blin  = (const float*)d_in[10];
    float* out = (float*)d_out;

    // LDS: 128KB weight staging + 2x4KB h ping-pong + tables (CDNA5 320KB/WGP)
    size_t smem = (size_t)(G_ * H_ * 2)          // wlds
                + (size_t)(2 * HSTRIDE * 2)      // hbuf ping-pong
                + (size_t)(G_ + G_ + H_ + 2 * ROWS + 128) * 4;
    hipLaunchKernelGGL(lstm_ae_kernel, dim3(B_ / ROWS), dim3(256), smem, stream,
                       x, Wih_e, Whh_e, bih_e, bhh_e,
                       Wih_d, Whh_d, bih_d, bhh_d, Wlin, blin, out);
}
